// MultiHead_27041114096095
// MI455X (gfx1250) — matrix operand order repaired and checked by reading, not compiled, not run on hardware
//
#include <hip/hip_runtime.h>

typedef _Float16 h16;
typedef __attribute__((ext_vector_type(16))) _Float16 v16h;
typedef __attribute__((ext_vector_type(8)))  _Float16 v8h;
typedef __attribute__((ext_vector_type(4)))  _Float16 v4h;
typedef __attribute__((ext_vector_type(8)))  float    v8f;
typedef __attribute__((ext_vector_type(4)))  float    v4f;

static constexpr int NB  = 16384;   // batches
static constexpr int S   = 50;      // sequence length
static constexpr int D   = 32;      // model dim
static constexpr int H   = 4;       // heads
static constexpr int SP  = 64;      // padded sequence (4 x 16 tiles)
static constexpr float EPSV   = 1e-5f;
static constexpr float QSCALE = 0.35355339059327379f;  // 1/sqrt(HD=8)

// ---------------- WMMA fragment helpers (A/B: contiguous k = 16h + s per lane half) ----------------

// A fragment (16x32 f16), LDS row-major (M x K), ld in halves (multiple of 8).
// lane<16: halves 0..15 <- K=k0..k0+15
// lane>=16: halves 0..15 <- K=k0+16..k0+31.
__device__ __forceinline__ v16h frag_A(const h16* base, int ld, int m0, int k0) {
  const int lane = threadIdx.x & 31;
  const int row  = m0 + (lane & 15);
  const int kh   = k0 + ((lane & 16) ? 16 : 0);
  const h16* p = base + row * ld + kh;
  v8h lo = *(const v8h*)(p);
  v8h hi = *(const v8h*)(p + 8);
  v16h r;
#pragma unroll
  for (int i = 0; i < 8; ++i) { r[i] = lo[i]; r[i + 8] = hi[i]; }
  return r;
}

// B fragment (32x16 f16) from a TRANSPOSED (N x K) row-major LDS buffer.
// lane<16: K=k0..k0+15 ; lane>=16: K=k0+16..k0+31, column = n0 + lane%16.
__device__ __forceinline__ v16h frag_B(const h16* baseT, int ld, int n0, int k0) {
  const int lane = threadIdx.x & 31;
  const int col  = n0 + (lane & 15);
  const int kk   = k0 + ((lane & 16) ? 16 : 0);
  const h16* p = baseT + col * ld + kk;
  v8h lo = *(const v8h*)(p);
  v8h hi = *(const v8h*)(p + 8);
  v16h r;
#pragma unroll
  for (int i = 0; i < 8; ++i) { r[i] = lo[i]; r[i + 8] = hi[i]; }
  return r;
}

// B fragment for per-head scores: Bt = k rows (seq x 32), keep only K in [8h, 8h+8).
__device__ __forceinline__ v16h frag_B_head(const h16* baseT, int ld, int n0, int h) {
  const int lane = threadIdx.x & 31;
  const int col  = n0 + (lane & 15);
  const int g    = (lane >> 4) & 1;          // this half-wave covers K window [16g,16g+16)
  v16h r;
#pragma unroll
  for (int i = 0; i < 16; ++i) r[i] = (h16)0.f;
  const int hk0 = h * 8;                      // head's global K base
  if ((hk0 >> 4) == g) {                      // head window lives in this half-wave's window
    const v8h lo = *(const v8h*)(baseT + col * ld + hk0);
    if (hk0 & 8) {
#pragma unroll
      for (int i = 0; i < 8; ++i) r[8 + i] = lo[i];
    } else {
#pragma unroll
      for (int i = 0; i < 8; ++i) r[i] = lo[i];
    }
  }
  return r;
}

__device__ __forceinline__ v8f wmma_f16(v16h a, v16h b, v8f c) {
  return __builtin_amdgcn_wmma_f32_16x16x32_f16(false, a, false, b, (short)0, c, false, false);
}

// Unconditional D-tile store as f16 (pad-row garbage is confined by construction).
__device__ __forceinline__ void store_D_f16(v8f d, h16* dst, int ld, int m0, int n0,
                                            float scale) {
  const int lane = threadIdx.x & 31;
  const int col  = n0 + (lane & 15);
  const int rb   = m0 + ((lane & 16) ? 8 : 0);
#pragma unroll
  for (int r = 0; r < 8; ++r) dst[(rb + r) * ld + col] = (h16)(d[r] * scale);
}

// Store D tile f32 into row-major LDS.
__device__ __forceinline__ void store_D_f32(v8f d, float* dst, int ld, int m0, int n0) {
  const int lane = threadIdx.x & 31;
  const int col  = n0 + (lane & 15);
  const int rb   = m0 + ((lane & 16) ? 8 : 0);
#pragma unroll
  for (int r = 0; r < 8; ++r) dst[(rb + r) * ld + col] = d[r];
}

// ---------------- fused LN + QKV proj + causal MHA + out proj + residual ----------------

__global__ __launch_bounds__(128) void mha_kernel(
    const float* __restrict__ Qin, const float* __restrict__ Kin, const float* __restrict__ Vin,
    const float* __restrict__ ln_g, const float* __restrict__ ln_b,
    const float* __restrict__ Wq, const float* __restrict__ bq,
    const float* __restrict__ Wk, const float* __restrict__ bk,
    const float* __restrict__ Wv, const float* __restrict__ bv,
    const float* __restrict__ Wo, const float* __restrict__ bo,
    float* __restrict__ out) {
  // LDS staging (~62 KB/block; 320 KB WGP pool -> ~5 blocks/WGP)
  __shared__ alignas(16) h16 xq[SP * D], xk[SP * D], xv[SP * D];       // LN'd inputs (f16)
  __shared__ alignas(16) h16 wqs[D * D], wks[D * D], wvs[D * D], wos[D * D];
  __shared__ alignas(16) h16 qb[SP * D];        // projected q (pre-scaled by 1/sqrt(hd))
  __shared__ alignas(16) h16 kb[SP * D];        // projected k
  __shared__ alignas(16) h16 vT[48 * SP];       // projected v, TRANSPOSED: vT[d][s]
  __shared__ alignas(16) h16 ctxb[SP * D];      // attention context
  __shared__ alignas(16) h16 Pb[SP * SP];       // softmax probs (f16, zero-padded)
  __shared__ alignas(16) float sb[SP * SP];     // raw scores (f32)

  const int b    = blockIdx.x;
  const int tid  = threadIdx.x;
  const int wave = tid >> 5;
  const int lane = tid & 31;
  const size_t base = (size_t)b * S * D;

  // Only vT rows 32..47 (head-3 N-tile overrun) need one-time zeroing;
  // the seq-pad columns (s >= 50) of rows 0..31 are zeroed by the select-store below.
  for (int i = tid; i < 16 * SP; i += 128) vT[32 * SP + i] = (h16)0.f;

  // Weights f32 -> f16 (row-major [n][k] == transposed-B layout for x @ W.T).
  for (int i = tid; i < D * D; i += 128) {
    wqs[i] = (h16)Wq[i]; wks[i] = (h16)Wk[i]; wvs[i] = (h16)Wv[i]; wos[i] = (h16)Wo[i];
  }

  // Shared LayerNorm on Q, K, V: one wave per row (lane = channel), butterfly reduce.
  for (int r = wave; r < 3 * S; r += 4) {
    const int t = r / S, m = r % S;
    const float* src = (t == 0) ? Qin : ((t == 1) ? Kin : Vin);
    h16* dst = (t == 0) ? xq : ((t == 1) ? xk : xv);
    float x = src[base + (size_t)m * D + lane];
    float s = x, s2 = x * x;
#pragma unroll
    for (int off = 16; off > 0; off >>= 1) {
      s  += __shfl_xor(s,  off, 32);
      s2 += __shfl_xor(s2, off, 32);
    }
    const float mu  = s * (1.f / 32.f);
    const float var = s2 * (1.f / 32.f) - mu * mu;
    const float y = (x - mu) * rsqrtf(var + EPSV) * ln_g[lane] + ln_b[lane];
    dst[m * D + lane] = (h16)y;
  }
  __syncthreads();

  // Q/K/V projections: 3 mats x (4 M-tiles x 2 N-tiles) = 24 WMMA jobs.
  for (int job = wave; job < 24; job += 4) {
    const int mat = job >> 3;
    const int t   = job & 7;
    const int mT  = t >> 1, nT = t & 1;
    const h16*  A  = (mat == 0) ? xq : ((mat == 1) ? xk : xv);
    const h16*  W  = (mat == 0) ? wqs : ((mat == 1) ? wks : wvs);
    const float* bias = (mat == 0) ? bq : ((mat == 1) ? bk : bv);
    v16h a  = frag_A(A, D, mT * 16, 0);
    v16h bb = frag_B(W, D, nT * 16, 0);
    const float b0 = bias[nT * 16 + (lane & 15)];
    v8f c;
#pragma unroll
    for (int r = 0; r < 8; ++r) c[r] = b0;
    v8f d = wmma_f16(a, bb, c);
    if (mat == 0) {
      store_D_f16(d, qb, D, mT * 16, nT * 16, QSCALE);   // pad rows confined (A operand)
    } else if (mat == 1) {
      store_D_f16(d, kb, D, mT * 16, nT * 16, 1.f);      // pad rows -> score cols >= 50, unread
    } else {
      // transpose select-store: vT[n][m], zeros for the seq-pad (K-dim must be clean)
      const int col = nT * 16 + (lane & 15);
      const int rb  = mT * 16 + ((lane & 16) ? 8 : 0);
#pragma unroll
      for (int r = 0; r < 8; ++r) {
        const int m = rb + r;
        vT[col * SP + m] = (m < S) ? (h16)d[r] : (h16)0.f;
      }
    }
  }
  __syncthreads();

  // Per-head: causal scores (lower-triangular tiles only) -> softmax -> context.
  for (int h = 0; h < H; ++h) {
    // scores: 10 of 16 tiles (nT <= mT). Loop index is wave-uniform -> EXEC all-ones.
    for (int t = wave; t < 16; t += 4) {
      const int mT = t >> 2, nT = t & 3;
      if (nT > mT) continue;
      v16h a  = frag_A(qb, D, mT * 16, 0);
      v16h bb = frag_B_head(kb, D, nT * 16, h);
      v8f c;
#pragma unroll
      for (int r = 0; r < 8; ++r) c[r] = 0.f;
      v8f d = wmma_f16(a, bb, c);
      store_D_f32(d, sb, SP, mT * 16, nT * 16);
    }
    __syncthreads();

    // Row softmax, vectorized. Pass 1: masked max. Pass 2: exp -> unnormalized f16 P.
    // Pass 3: rescale P by 1/sum (b128 LDS traffic throughout).
    for (int m = tid; m < S; m += 128) {
      const float* row = sb + m * SP;
      float mx = -3.0e38f;
#pragma unroll
      for (int i = 0; i < 16; ++i) {
        v4f s = *(const v4f*)(row + 4 * i);
#pragma unroll
        for (int j = 0; j < 4; ++j) {
          const int n = 4 * i + j;
          mx = fmaxf(mx, (n <= m) ? s[j] : -3.0e38f);
        }
      }
      float sum = 0.f;
      h16* prow = Pb + m * SP;
#pragma unroll
      for (int i = 0; i < 16; ++i) {
        v4f s = *(const v4f*)(row + 4 * i);
        v4h e;
#pragma unroll
        for (int j = 0; j < 4; ++j) {
          const int n = 4 * i + j;
          const float ev = (n <= m) ? __expf(s[j] - mx) : 0.f;
          sum += ev;
          e[j] = (h16)ev;
        }
        *(v4h*)(prow + 4 * i) = e;
      }
      const float inv = 1.f / sum;
#pragma unroll
      for (int i = 0; i < 8; ++i) {
        v8h p = *(const v8h*)(prow + 8 * i);
#pragma unroll
        for (int j = 0; j < 8; ++j) p[j] = (h16)((float)p[j] * inv);
        *(v8h*)(prow + 8 * i) = p;
      }
    }
    __syncthreads();

    // context: ctx_h = P (50x64, K padded w/ zeros) x vT rows [8h,8h+8) ; 4 M-tiles.
    for (int mT = wave; mT < 4; mT += 4) {
      v8f c;
#pragma unroll
      for (int r = 0; r < 8; ++r) c[r] = 0.f;
#pragma unroll
      for (int ks = 0; ks < 2; ++ks) {
        v16h a  = frag_A(Pb, SP, mT * 16, ks * 32);
        v16h bb = frag_B(vT, SP, h * 8, ks * 32);
        c = wmma_f16(a, bb, c);
      }
      const int col = lane & 15;
      const int rb  = mT * 16 + ((lane & 16) ? 8 : 0);
      if (col < 8) {
#pragma unroll
        for (int r = 0; r < 8; ++r) ctxb[(rb + r) * D + h * 8 + col] = (h16)c[r];
      }
    }
    __syncthreads();
  }

  // Output projection + bias + residual. Edge masking only for the mT==3 tile
  // (wave-uniform branch -> no per-element exec thrash on the 3 full tiles).
  for (int job = wave; job < 8; job += 4) {
    const int mT = job >> 1, nT = job & 1;
    v16h a  = frag_A(ctxb, D, mT * 16, 0);
    v16h bb = frag_B(wos, D, nT * 16, 0);
    const float b0 = bo[nT * 16 + (lane & 15)];
    v8f c;
#pragma unroll
    for (int r = 0; r < 8; ++r) c[r] = b0;
    v8f d = wmma_f16(a, bb, c);
    const int col = nT * 16 + (lane & 15);
    const int rb  = mT * 16 + ((lane & 16) ? 8 : 0);
    if (mT < 3) {
#pragma unroll
      for (int r = 0; r < 8; ++r) {
        const size_t idx = base + (size_t)(rb + r) * D + col;
        out[idx] = d[r] + Qin[idx];   // residual uses pre-LN Q
      }
    } else {
#pragma unroll
      for (int r = 0; r < 8; ++r) {
        const int m = rb + r;
        if (m < S) {
          const size_t idx = base + (size_t)m * D + col;
          out[idx] = d[r] + Qin[idx];
        }
      }
    }
  }
}

extern "C" void kernel_launch(void* const* d_in, const int* in_sizes, int n_in,
                              void* d_out, int out_size, void* d_ws, size_t ws_size,
                              hipStream_t stream) {
  const float* Q    = (const float*)d_in[0];
  const float* K    = (const float*)d_in[1];
  const float* V    = (const float*)d_in[2];
  // d_in[3] = boolean causal mask: implemented analytically in-kernel, unused here.
  const float* ln_g = (const float*)d_in[4];
  const float* ln_b = (const float*)d_in[5];
  const float* Wq   = (const float*)d_in[6];
  const float* bq   = (const float*)d_in[7];
  const float* Wk   = (const float*)d_in[8];
  const float* bk   = (const float*)d_in[9];
  const float* Wv   = (const float*)d_in[10];
  const float* bv   = (const float*)d_in[11];
  const float* Wo   = (const float*)d_in[12];
  const float* bo   = (const float*)d_in[13];
  float* out = (float*)d_out;

  dim3 grid(NB), block(128);
  mha_kernel<<<grid, block, 0, stream>>>(Q, K, V, ln_g, ln_b,
                                         Wq, bq, Wk, bk, Wv, bv, Wo, bo, out);
}
